// MultiHead_27608049778912
// MI455X (gfx1250) — matrix operand order repaired and checked by reading, not compiled, not run on hardware
//
#include <hip/hip_runtime.h>

// ---------------------------------------------------------------------------
// Multi-head causal attention for MI455X (gfx1250, wave32, WMMA bf16)
//   B=8, T=2048, C=384, H=6, D=64
// Pipeline: [qkv_proj (WMMA)] -> [flash attention (WMMA)] -> [out_proj (WMMA)]
// bf16 WMMA / f32 accumulate; async global->LDS staging; v_cvt_pk_bf16_f32.
// ---------------------------------------------------------------------------

#define kB 8
#define kT 2048
#define kC 384
#define kH 6
#define kD 64

typedef __attribute__((ext_vector_type(16))) __bf16 v16bf;
typedef __attribute__((ext_vector_type(2)))  __bf16 v2bf;
typedef __attribute__((ext_vector_type(8)))  float  v8f;
typedef __attribute__((ext_vector_type(4)))  unsigned int u32x4;
typedef __attribute__((ext_vector_type(4)))  float  f32x4;

union FragB16 {
    v16bf v;
    u32x4 q[2];
    unsigned int u[8];
    unsigned short s[16];
};

// pack two f32 -> two bf16 in one uint (single v_cvt_pk_bf16_f32)
static __device__ __forceinline__ unsigned int f2bf2(float a, float b) {
#if __has_builtin(__builtin_amdgcn_cvt_pk_bf16_f32)
    union { v2bf v; unsigned int u; } c;
    c.v = __builtin_amdgcn_cvt_pk_bf16_f32(a, b);
    return c.u;
#else
    unsigned int r;
    asm("v_cvt_pk_bf16_f32 %0, %1, %2" : "=v"(r) : "v"(a), "v"(b));
    return r;
#endif
}

static __device__ __forceinline__ v8f v8f_zero() {
    v8f z;
#pragma unroll
    for (int i = 0; i < 8; ++i) z[i] = 0.0f;
    return z;
}

// Async global->LDS 16B copy (gfx1250 GLOBAL_LOAD_ASYNC_TO_LDS_B128, ASYNCcnt).
// ldsOff = wave-relative LDS byte offset (low 32 bits of generic shared addr).
static __device__ __forceinline__ void async_copy_b128(void* ldsPtr, const void* gPtr) {
    unsigned int ldsOff = (unsigned int)(size_t)ldsPtr;
    asm volatile("global_load_async_to_lds_b128 %0, %1, off"
                 :: "v"(ldsOff), "v"(gPtr) : "memory");
}
static __device__ __forceinline__ void async_wait0() {
    asm volatile("s_wait_asynccnt 0x0" ::: "memory");
}

// A-matrix fragment (16x32 bf16, MxK). lane: m = lane&15.
// VGPR i holds K pair: k = kbase + (laneHi?16:0) + i*2 + {0,1}
static __device__ __forceinline__ v16bf load_a_frag(const unsigned short* rowPtr,
                                                    int kbase, int laneHi) {
    FragB16 f;
#pragma unroll
    for (int i = 0; i < 8; ++i) {
        int k = kbase + (laneHi ? 16 : 0) + (i << 1);
        f.u[i] = *(const unsigned int*)(rowPtr + k);
    }
    return f.v;
}

// B-matrix fragment (32x16 bf16, KxN). lane: n = lane&15, holds K range
// [laneHi*16, laneHi*16+16) contiguous; p -> those 16 bf16 (32B aligned).
static __device__ __forceinline__ v16bf load_b_contig(const unsigned short* p) {
    FragB16 f;
    f.q[0] = *(const u32x4*)(p);
    f.q[1] = *(const u32x4*)(p + 8);
    return f.v;
}

static __device__ __forceinline__ v8f wmma_bf16(v16bf a, v16bf b, v8f c) {
    return __builtin_amdgcn_wmma_f32_16x16x32_bf16(false, a, false, b,
                                                   (short)0, c, false, false);
}

static __device__ __forceinline__ float half16_max(float v) {
    v = fmaxf(v, __shfl_xor(v, 1, 32));
    v = fmaxf(v, __shfl_xor(v, 2, 32));
    v = fmaxf(v, __shfl_xor(v, 4, 32));
    v = fmaxf(v, __shfl_xor(v, 8, 32));
    return v;
}
static __device__ __forceinline__ float half16_sum(float v) {
    v += __shfl_xor(v, 1, 32);
    v += __shfl_xor(v, 2, 32);
    v += __shfl_xor(v, 4, 32);
    v += __shfl_xor(v, 8, 32);
    return v;
}

// ---------------------------------------------------------------------------
// Kernel 1: QKV projection.  grid (T/128, B*H, 3), block 256 (8 waves).
// ---------------------------------------------------------------------------
__global__ __launch_bounds__(256)
void mha_qkv_proj(const float* __restrict__ x,
                  const float* __restrict__ Wq,
                  const float* __restrict__ Wk,
                  const float* __restrict__ Wv,
                  unsigned short* __restrict__ qkv) {
    __shared__ unsigned short xs[128 * 32];   // x tile, [row][c] bf16
    __shared__ unsigned short wsm[64 * 32];   // W tile transposed, [d][c] bf16

    const int tTile = blockIdx.x;
    const int bh    = blockIdx.y;
    const int which = blockIdx.z;
    const int h = bh % kH;

    const float* W = (which == 0) ? Wq : ((which == 1) ? Wk : Wv);
    W += (size_t)h * kC * kD;
    const float* xrow = x + ((size_t)(bh / kH) * kT + (size_t)tTile * 128) * kC;

    const int tid = threadIdx.x;
    const int lane = tid & 31, wave = tid >> 5;
    const int laneHi = lane >> 4, laneLo = lane & 15;

    v8f acc[4];
#pragma unroll
    for (int nb = 0; nb < 4; ++nb) acc[nb] = v8f_zero();

    for (int c0 = 0; c0 < kC; c0 += 32) {
        __syncthreads();
        {   // stage x tile: 128x32 f32 -> bf16, packed b32 LDS stores
            int idx = tid * 16;
            int row = idx >> 5, cc = idx & 31;
            const float* src = xrow + (size_t)row * kC + c0 + cc;
            f32x4 f0 = *(const f32x4*)(src);
            f32x4 f1 = *(const f32x4*)(src + 4);
            f32x4 f2 = *(const f32x4*)(src + 8);
            f32x4 f3 = *(const f32x4*)(src + 12);
            float tmp[16] = {f0.x,f0.y,f0.z,f0.w, f1.x,f1.y,f1.z,f1.w,
                             f2.x,f2.y,f2.z,f2.w, f3.x,f3.y,f3.z,f3.w};
            unsigned int* dst32 = (unsigned int*)&xs[row * 32 + cc];
#pragma unroll
            for (int e = 0; e < 8; ++e) dst32[e] = f2bf2(tmp[2*e], tmp[2*e+1]);
        }
        {   // stage W tile transposed: read [c][d] coalesced, write [d][c]
            int idx = tid * 8;                 // 32*64/256 = 8 per thread
            int cl = idx >> 6, d0 = idx & 63;
            const float* src = W + (size_t)(c0 + cl) * kD + d0;
            f32x4 f0 = *(const f32x4*)(src);
            f32x4 f1 = *(const f32x4*)(src + 4);
            float tmp[8] = {f0.x,f0.y,f0.z,f0.w, f1.x,f1.y,f1.z,f1.w};
#pragma unroll
            for (int e = 0; e < 4; ++e) {
                unsigned int pk = f2bf2(tmp[2*e], tmp[2*e+1]);
                wsm[(d0 + 2*e)     * 32 + cl] = (unsigned short)pk;
                wsm[(d0 + 2*e + 1) * 32 + cl] = (unsigned short)(pk >> 16);
            }
        }
        __syncthreads();

        const unsigned short* arow = &xs[(wave * 16 + laneLo) * 32];
        v16bf a = load_a_frag(arow, 0, laneHi);
        v16bf bb[4];
#pragma unroll
        for (int nb = 0; nb < 4; ++nb)
            bb[nb] = load_b_contig(&wsm[(nb * 16 + laneLo) * 32 + (laneHi ? 16 : 0)]);
#pragma unroll
        for (int nb = 0; nb < 4; ++nb)
            acc[nb] = wmma_bf16(a, bb[nb], acc[nb]);
    }

    // fold attention scale C^-0.5 into Q
    const float qscale = (which == 0) ? rsqrtf((float)kC) : 1.0f;
    const size_t QCNT = (size_t)kB * kH * kT * kD;
    unsigned short* dst = qkv + (size_t)which * QCNT +
                          ((size_t)bh * kT + (size_t)tTile * 128 + wave * 16) * kD;
#pragma unroll
    for (int nb = 0; nb < 4; ++nb) {
        int col = nb * 16 + laneLo;
#pragma unroll
        for (int r = 0; r < 4; ++r) {
            unsigned int pk = f2bf2(acc[nb][2*r] * qscale, acc[nb][2*r+1] * qscale);
            int row = 2*r + (laneHi ? 8 : 0);
            dst[(size_t)row * kD + col]       = (unsigned short)pk;
            dst[(size_t)(row + 1) * kD + col] = (unsigned short)(pk >> 16);
        }
    }
}

// ---------------------------------------------------------------------------
// Kernel 2: causal flash attention. grid (T/128, B*H), block 256 (8 waves).
// ---------------------------------------------------------------------------
__global__ __launch_bounds__(256)
void mha_flash_attn(const unsigned short* __restrict__ qkv,
                    unsigned short* __restrict__ oh) {
    __shared__ unsigned short ks[64 * 64];        // K tile [kv][d]
    __shared__ unsigned short vsT[64 * 64];       // V tile transposed [d][kv]
    __shared__ unsigned short pbuf[8 * 16 * 64];  // per-wave P staging

    const int qTile = blockIdx.x;
    const int bh    = blockIdx.y;
    const int tid = threadIdx.x;
    const int lane = tid & 31, wave = tid >> 5;
    const int laneHi = lane >> 4, laneLo = lane & 15;

    const size_t QCNT = (size_t)kB * kH * kT * kD;
    const unsigned short* qb = qkv + (size_t)bh * kT * kD;
    const unsigned short* kb = qb + QCNT;
    const unsigned short* vb = qb + 2 * QCNT;

    const int qBase = qTile * 128;
    const int myRow0 = qBase + wave * 16;

    // Q fragments for this wave's 16 rows (D=64 -> two K=32 A-frags)
    v16bf aq[2];
    {
        const unsigned short* qrow = qb + (size_t)(myRow0 + laneLo) * kD;
        aq[0] = load_a_frag(qrow, 0, laneHi);
        aq[1] = load_a_frag(qrow, 32, laneHi);
    }

    float mst[8], lst[8];
    v8f o[4];
#pragma unroll
    for (int r = 0; r < 8; ++r) { mst[r] = -3.0e38f; lst[r] = 0.0f; }
#pragma unroll
    for (int nb = 0; nb < 4; ++nb) o[nb] = v8f_zero();

    const int nTiles = qBase / 64 + 2;   // causal: keys up to qBase+127
    for (int kt = 0; kt < nTiles; ++kt) {
        const int kvBase = kt * 64;
        __syncthreads();
        {   // stage K tile via async global->LDS (2 x b128 per thread)
            const unsigned short* src = kb + (size_t)kvBase * kD;
            int o16 = tid * 16;
            async_copy_b128(&ks[o16],     &src[o16]);
            async_copy_b128(&ks[o16 + 8], &src[o16 + 8]);
        }
        {   // stage V tile transposed (manual: transpose needs data movement)
            const unsigned short* src = vb + (size_t)kvBase * kD;
            int o16 = tid * 16;
            int kv = o16 >> 6, d0 = o16 & 63;
            u32x4 p0 = *(const u32x4*)&src[o16];
            u32x4 p1 = *(const u32x4*)&src[o16 + 8];
            unsigned short tmp[16];
            *(u32x4*)&tmp[0] = p0;
            *(u32x4*)&tmp[8] = p1;
#pragma unroll
            for (int e = 0; e < 16; ++e) vsT[(d0 + e) * 64 + kv] = tmp[e];
        }
        async_wait0();
        __syncthreads();

        // skip compute on fully-masked tiles (wave-uniform; no barriers inside)
        if (kvBase > myRow0 + 15) continue;

        // S = Q * K^T   (4 col-blocks x 2 K-steps = 8 WMMA)
        v8f s[4];
#pragma unroll
        for (int nb = 0; nb < 4; ++nb) s[nb] = v8f_zero();
#pragma unroll
        for (int k2 = 0; k2 < 2; ++k2) {
            v16bf bk[4];
#pragma unroll
            for (int nb = 0; nb < 4; ++nb)
                bk[nb] = load_b_contig(
                    &ks[(nb * 16 + laneLo) * 64 + k2 * 32 + (laneHi ? 16 : 0)]);
#pragma unroll
            for (int nb = 0; nb < 4; ++nb)
                s[nb] = wmma_bf16(aq[k2], bk[nb], s[nb]);
        }

        // causal mask (diagonal tiles only; wave-uniform test)
        if (kvBase + 63 > myRow0) {
#pragma unroll
            for (int r = 0; r < 8; ++r) {
                int rowG = myRow0 + r + laneHi * 8;
#pragma unroll
                for (int nb = 0; nb < 4; ++nb) {
                    int colG = kvBase + nb * 16 + laneLo;
                    if (colG > rowG) s[nb][r] = -3.0e38f;
                }
            }
        }

        // per-row max + online softmax update
        float alpha[8];
#pragma unroll
        for (int r = 0; r < 8; ++r) {
            float mx = fmaxf(fmaxf(s[0][r], s[1][r]), fmaxf(s[2][r], s[3][r]));
            mx = half16_max(mx);
            float mnew = fmaxf(mst[r], mx);
            alpha[r] = __expf(mst[r] - mnew);
            mst[r] = mnew;
            float sum = 0.0f;
#pragma unroll
            for (int nb = 0; nb < 4; ++nb) {
                float p = __expf(s[nb][r] - mnew);
                s[nb][r] = p;
                sum += p;
            }
            lst[r] = lst[r] * alpha[r] + half16_sum(sum);
        }
#pragma unroll
        for (int nb = 0; nb < 4; ++nb)
#pragma unroll
            for (int r = 0; r < 8; ++r) o[nb][r] *= alpha[r];

        // P -> bf16 via per-wave LDS (C-layout write, A-layout reload)
        unsigned short* pw = &pbuf[wave * 16 * 64];
#pragma unroll
        for (int nb = 0; nb < 4; ++nb) {
            int col = nb * 16 + laneLo;
#pragma unroll
            for (int r = 0; r < 4; ++r) {
                unsigned int pk = f2bf2(s[nb][2*r], s[nb][2*r+1]);
                int m = 2*r + laneHi * 8;
                pw[m * 64 + col]       = (unsigned short)pk;
                pw[(m + 1) * 64 + col] = (unsigned short)(pk >> 16);
            }
        }

        // O += P * V   (8 WMMA)
#pragma unroll
        for (int k2 = 0; k2 < 2; ++k2) {
            v16bf ap = load_a_frag(&pw[laneLo * 64], k2 * 32, laneHi);
            v16bf bv[4];
#pragma unroll
            for (int nb = 0; nb < 4; ++nb)
                bv[nb] = load_b_contig(
                    &vsT[(nb * 16 + laneLo) * 64 + k2 * 32 + (laneHi ? 16 : 0)]);
#pragma unroll
            for (int nb = 0; nb < 4; ++nb)
                o[nb] = wmma_bf16(ap, bv[nb], o[nb]);
        }
    }

    // normalize and store head output (bf16, [B][T][C] with head offset)
    const int b = bh / kH, h = bh % kH;
    unsigned short* dst = oh + ((size_t)b * kT + myRow0) * kC + h * kD;
#pragma unroll
    for (int r = 0; r < 8; ++r) { lst[r] = 1.0f / lst[r]; }
#pragma unroll
    for (int nb = 0; nb < 4; ++nb) {
        int col = nb * 16 + laneLo;
#pragma unroll
        for (int r = 0; r < 4; ++r) {
            unsigned int pk = f2bf2(o[nb][2*r] * lst[2*r], o[nb][2*r+1] * lst[2*r+1]);
            int row = 2*r + laneHi * 8;
            dst[(size_t)row * kC + col]       = (unsigned short)pk;
            dst[(size_t)(row + 1) * kC + col] = (unsigned short)(pk >> 16);
        }
    }
}

// ---------------------------------------------------------------------------
// Kernel 3: output projection y = oh @ Wp^T + bp. grid (B*T/128, C/64).
// ---------------------------------------------------------------------------
__global__ __launch_bounds__(256)
void mha_out_proj(const unsigned short* __restrict__ oh,
                  const float* __restrict__ Wp,
                  const float* __restrict__ bias,
                  float* __restrict__ y) {
    __shared__ unsigned short as_[128 * 32];  // activation tile [row][ci]
    __shared__ unsigned short ws_[64 * 32];   // Wp tile [co][ci] bf16

    const int rTile = blockIdx.x;
    const int cTile = blockIdx.y;
    const int tid = threadIdx.x;
    const int lane = tid & 31, wave = tid >> 5;
    const int laneHi = lane >> 4, laneLo = lane & 15;
    const int rBase = rTile * 128, coBase = cTile * 64;

    v8f acc[4];
#pragma unroll
    for (int nb = 0; nb < 4; ++nb) acc[nb] = v8f_zero();

    for (int c0 = 0; c0 < kC; c0 += 32) {
        __syncthreads();
        {   // stage activations (already bf16) via async global->LDS
            int o16 = tid * 16;
            int row = o16 >> 5, cc = o16 & 31;
            const unsigned short* src = oh + (size_t)(rBase + row) * kC + c0 + cc;
            async_copy_b128(&as_[o16],     src);
            async_copy_b128(&as_[o16 + 8], src + 8);
        }
        {   // stage Wp tile: rows contiguous in ci -> packed b32 LDS stores
            int idx = tid * 8;
            int co = idx >> 5, ci0 = idx & 31;
            const float* src = Wp + (size_t)(coBase + co) * kC + c0 + ci0;
            f32x4 f0 = *(const f32x4*)(src);
            f32x4 f1 = *(const f32x4*)(src + 4);
            float tmp[8] = {f0.x,f0.y,f0.z,f0.w, f1.x,f1.y,f1.z,f1.w};
            unsigned int* d32 = (unsigned int*)&ws_[co * 32 + ci0];
#pragma unroll
            for (int e = 0; e < 4; ++e) d32[e] = f2bf2(tmp[2*e], tmp[2*e+1]);
        }
        async_wait0();
        __syncthreads();

        v16bf a = load_a_frag(&as_[(wave * 16 + laneLo) * 32], 0, laneHi);
        v16bf bb[4];
#pragma unroll
        for (int nb = 0; nb < 4; ++nb)
            bb[nb] = load_b_contig(&ws_[(nb * 16 + laneLo) * 32 + (laneHi ? 16 : 0)]);
#pragma unroll
        for (int nb = 0; nb < 4; ++nb)
            acc[nb] = wmma_bf16(a, bb[nb], acc[nb]);
    }

#pragma unroll
    for (int nb = 0; nb < 4; ++nb) {
        int co = coBase + nb * 16 + laneLo;
        float bv = bias[co];
#pragma unroll
        for (int r = 0; r < 8; ++r) {
            int row = rBase + wave * 16 + r + laneHi * 8;
            y[(size_t)row * kC + co] = acc[nb][r] + bv;
        }
    }
}

// ---------------------------------------------------------------------------
extern "C" void kernel_launch(void* const* d_in, const int* in_sizes, int n_in,
                              void* d_out, int out_size, void* d_ws, size_t ws_size,
                              hipStream_t stream) {
    const float* x  = (const float*)d_in[0];
    const float* Wq = (const float*)d_in[1];
    const float* Wk = (const float*)d_in[2];
    const float* Wv = (const float*)d_in[3];
    const float* Wp = (const float*)d_in[4];
    const float* bp = (const float*)d_in[5];
    float* out = (float*)d_out;

    // workspace layout (bf16/ushort): q | k | v | head-concat output (~50 MB)
    unsigned short* ws = (unsigned short*)d_ws;
    const size_t QCNT = (size_t)kB * kH * kT * kD;   // 6,291,456 elems each
    unsigned short* qkv = ws;                        // 3*QCNT
    unsigned short* oh  = ws + 3 * QCNT;             // kB*kT*kC

    mha_qkv_proj<<<dim3(kT / 128, kB * kH, 3), 256, 0, stream>>>(x, Wq, Wk, Wv, qkv);
    mha_flash_attn<<<dim3(kT / 128, kB * kH), 256, 0, stream>>>(qkv, oh);
    mha_out_proj<<<dim3((kB * kT) / 128, kC / 64), 256, 0, stream>>>(oh, Wp, bp, out);
}
